// Decoder_34626026341080
// MI455X (gfx1250) — matrix operand order repaired and checked by reading, not compiled, not run on hardware
//
#include <hip/hip_runtime.h>

#define NB    256
#define NTH   256
#define S_LEN 1024
#define ENC_D 512
#define HID   1024
#define ATT   128
#define VOC   32000
#define T_LEN 512

typedef __bf16 bf16;
typedef __attribute__((ext_vector_type(16))) __bf16 v16bf;
typedef __attribute__((ext_vector_type(8)))  __bf16 v8bf;
typedef __attribute__((ext_vector_type(8)))  float  v8f;

struct Params {
  // inputs
  const float *enc, *h0, *c0, *emb, *Wa_enc, *Wa_h, *v_a;
  const float *W_ih, *W_hh, *b_ih, *b_hh, *W_out, *b_out;
  const int *sos, *tgt;
  // outputs
  float *out_logits, *out_attn;
  // workspace
  unsigned *sync;
  float *h, *c, *wexp, *partialZ, *enc_proj, *Wa_hT;
  bf16 *h_bf, *x_bf, *hnb0, *hnb1;
  bf16 *Wih_bf, *Whh_bf, *Wout_bf;   // Wih/Whh stored gate-row-PERMUTED (see prologue)
};

__device__ __forceinline__ void grid_barrier(unsigned* cnt) {
  __syncthreads();
  if (threadIdx.x == 0) {
    __threadfence();
    unsigned prev = __hip_atomic_fetch_add(cnt, 1u, __ATOMIC_ACQ_REL, __HIP_MEMORY_SCOPE_AGENT);
    unsigned target = (prev / NB + 1u) * NB;
    while (__hip_atomic_load(cnt, __ATOMIC_ACQUIRE, __HIP_MEMORY_SCOPE_AGENT) < target) {
      __builtin_amdgcn_s_sleep(2);
    }
  }
  __syncthreads();
}

__device__ __forceinline__ float sigmoidf_(float x) { return 1.0f / (1.0f + __expf(-x)); }

// 16 output rows [r0, r0+16) of W(row-major, rowstride K) times vector x, via WMMA.
// A tile: 16x32 bf16 per CDNA5 layout (lane<16: K 0..7 / 16..23, lane>=16: K 8..15 / 24..31).
// B tile: 32x16, same K layout as A, every column = x (lane<16: K 0..7 / 16..23, lane>=16: K 8..15 / 24..31).
// Two k-interleaved accumulator chains hide the WMMA->WMMA dependency delay.
__device__ __forceinline__ v8f wmma_matvec(const bf16* __restrict__ W,
                                           const bf16* __restrict__ x,
                                           int K, int r0, int lane, v8f acc_in) {
  const int m    = lane & 15;
  const int koff = (lane & 16) ? 8 : 0;
  const bf16* rowp = W + (size_t)(r0 + m) * (size_t)K + koff;
  const bf16* xs = x + koff;
  v8f acc0 = acc_in;
  v8f acc1 = {};
  union Bf { v16bf v; unsigned u[8]; v8bf h[2]; };
  Bf b0, b1;
  #pragma unroll
  for (int i = 0; i < 8; ++i) { b0.u[i] = 0u; b1.u[i] = 0u; }
  #pragma unroll 2
  for (int k0 = 0; k0 < K; k0 += 64) {
    union { v16bf v; v8bf h[2]; } a0, a1;
    a0.h[0] = *(const v8bf*)(rowp + k0);             // global_load_b128
    a0.h[1] = *(const v8bf*)(rowp + k0 + 16);        // global_load_b128
    a1.h[0] = *(const v8bf*)(rowp + k0 + 32);
    a1.h[1] = *(const v8bf*)(rowp + k0 + 48);
    b0.h[0] = *(const v8bf*)(xs + k0);
    b0.h[1] = *(const v8bf*)(xs + k0 + 16);
    b1.h[0] = *(const v8bf*)(xs + k0 + 32);
    b1.h[1] = *(const v8bf*)(xs + k0 + 48);
    acc0 = __builtin_amdgcn_wmma_f32_16x16x32_bf16(false, a0.v, false, b0.v,
                                                   (short)0, acc0, false, false);
    acc1 = __builtin_amdgcn_wmma_f32_16x16x32_bf16(false, a1.v, false, b1.v,
                                                   (short)0, acc1, false, false);
  }
  #pragma unroll
  for (int i = 0; i < 8; ++i) acc0[i] += acc1[i];
  return acc0;
}

__global__ void __launch_bounds__(NTH, 1) decoder_persistent(Params p) {
  const int tid  = threadIdx.x;
  const int b    = blockIdx.x;
  const int gtid = b * NTH + tid;
  const int lane = tid & 31;
  const int wave = tid >> 5;
  const size_t NTOT = (size_t)NB * NTH;

  __shared__ float s_vec[1024];   // staged h (P1) / wexp (P2)
  __shared__ float s_ah[ATT];
  __shared__ float s_red[NTH];
  __shared__ float s_w[32];
  __shared__ float s_Z;

  // ================= Prologue (once per call) =================
  // Gate weights: convert to bf16 AND permute rows so that destination tile row
  // nr = w*16 + m  holds source row (m&3)*HID + w*4 + (m>>2).
  // => after WMMA, wave w / lane 0 holds {i,f,g,o} for dims w*4+{0,1},
  //    lane 16 holds {i,f,g,o} for dims w*4+{2,3}: the LSTM cell fuses in-wave.
  for (size_t i = gtid; i < (size_t)4 * HID * HID; i += NTOT) {
    size_t nr = i >> 10; int col = (int)(i & 1023);
    int w = (int)(nr >> 4), m = (int)(nr & 15);
    size_t sr = (size_t)(m & 3) * HID + (size_t)w * 4 + (size_t)(m >> 2);
    p.Wih_bf[i] = (bf16)p.W_ih[sr * HID + col];
  }
  for (size_t i = gtid; i < (size_t)4 * HID * HID; i += NTOT) {
    size_t nr = i >> 10; int col = (int)(i & 1023);
    int w = (int)(nr >> 4), m = (int)(nr & 15);
    size_t sr = (size_t)(m & 3) * HID + (size_t)w * 4 + (size_t)(m >> 2);
    p.Whh_bf[i] = (bf16)p.W_hh[sr * HID + col];
  }
  for (size_t i = gtid; i < (size_t)VOC * HID; i += NTOT) p.Wout_bf[i] = (bf16)p.W_out[i];
  for (size_t i = gtid; i < (size_t)ATT * HID; i += NTOT) {     // transpose Wa_h -> (K, A)
    int a = (int)(i >> 10), k = (int)(i & 1023);
    p.Wa_hT[(size_t)k * ATT + a] = p.Wa_h[i];
  }
  for (size_t i = gtid; i < HID; i += NTOT) { p.h[i] = p.h0[i]; p.c[i] = p.c0[i]; }
  for (size_t o = gtid; o < (size_t)S_LEN * ATT; o += NTOT) {   // enc_proj[s*128+a]
    int s = (int)(o >> 7), a = (int)(o & 127);
    const float4* er = (const float4*)(p.enc    + (size_t)s * ENC_D);
    const float4* wr = (const float4*)(p.Wa_enc + (size_t)a * ENC_D);
    float acc = 0.f;
    for (int k = 0; k < ENC_D / 4; ++k) {
      float4 e = er[k], w = wr[k];
      acc += e.x * w.x + e.y * w.y + e.z * w.z + e.w * w.w;
    }
    p.enc_proj[o] = acc;
  }
  grid_barrier(p.sync);

  // ================= Sequential decode =================
  for (int t = 0; t < T_LEN; ++t) {
    const int tok = (t == 0) ? p.sos[0] : p.tgt[t - 1];

    // ---- P1: scores + exp (block-local ah), h->bf16, emb->x ----
    if (b < 32) {
      for (int i = tid; i < HID; i += NTH) s_vec[i] = p.h[i];   // stage h in LDS
      __syncthreads();
      { // ah = Wa_h @ h, redundantly per block (coalesced via Wa_hT)
        int a = tid & 127, half = tid >> 7;
        float acc = 0.f;
        for (int k = half * 512; k < half * 512 + 512; ++k)
          acc += p.Wa_hT[(size_t)k * ATT + a] * s_vec[k];
        s_red[tid] = acc;
      }
      __syncthreads();
      if (tid < ATT) s_ah[tid] = s_red[tid] + s_red[tid + 128];
      __syncthreads();
      { // score_s = v_a . tanh(enc_proj[s] + ah), 32 scores/block, 8 partials each
        int sl = tid >> 3, part = tid & 7;
        int s = b * 32 + sl;
        const float* ep = p.enc_proj + (size_t)s * ATT;
        float acc = 0.f;
        for (int a = 0; a < 16; ++a) {
          int ga = part * 16 + a;
          acc += p.v_a[ga] * tanhf(ep[ga] + s_ah[ga]);
        }
        s_red[sl * 8 + part] = acc;
      }
      __syncthreads();
      if (tid < 32) {
        float sc = 0.f;
        for (int j = 0; j < 8; ++j) sc += s_red[tid * 8 + j];
        float w = __expf(sc);          // scores tanh-bounded -> safe without max-sub
        p.wexp[b * 32 + tid] = w;
        s_w[tid] = w;
      }
      __syncthreads();
      if (tid == 0) {
        float z = 0.f;
        for (int j = 0; j < 32; ++j) z += s_w[j];
        p.partialZ[b] = z;
      }
    } else if (b == 32) {
      for (int i = tid; i < HID; i += NTH) p.h_bf[i] = (bf16)p.h[i];
    } else if (b == 33) {
      const float* er = p.emb + (size_t)tok * 512;
      for (int i = tid; i < 512; i += NTH) p.x_bf[i] = (bf16)er[i];
    }
    grid_barrier(p.sync);

    // ---- P2: context (-> x_bf[512:1024]) and attention output ----
    if (b < 16) {
      for (int i = tid; i < S_LEN; i += NTH) s_vec[i] = p.wexp[i];  // stage weights
      if (tid == 0) {
        float z = 0.f;
        for (int j = 0; j < 32; ++j) z += p.partialZ[j];
        s_Z = z;
      }
      __syncthreads();
      int jl = tid & 31, sg = tid >> 5;
      int j = b * 32 + jl;
      float acc = 0.f;
      for (int s = sg * 128; s < sg * 128 + 128; ++s)
        acc += s_vec[s] * p.enc[(size_t)s * ENC_D + j];
      s_red[sg * 32 + jl] = acc;
      __syncthreads();
      if (tid < 32) {
        float cx = 0.f;
        for (int g = 0; g < 8; ++g) cx += s_red[g * 32 + tid];
        p.x_bf[512 + b * 32 + tid] = (bf16)(cx / s_Z);
      }
    } else if (b < 48) {
      if (tid == 0) {
        float z = 0.f;
        for (int j = 0; j < 32; ++j) z += p.partialZ[j];
        s_Z = z;
      }
      __syncthreads();
      if (tid < 32) {
        int s = (b - 16) * 32 + tid;
        p.out_attn[(size_t)t * S_LEN + s] = p.wexp[s] / s_Z;
      }
    }
    grid_barrier(p.sync);

    // ---- P3: gates (WMMA over permuted rows) + fused LSTM cell ----
    if (b < 32) {
      const int w  = b * 8 + wave;       // 256 waves, 4 hidden dims each
      const int r0 = w * 16;             // permuted-row tile base
      const int d0 = w * 4;              // hidden dims [d0, d0+4)
      v8f acc = {};
      acc = wmma_matvec(p.Wih_bf, p.x_bf, HID, r0, lane, acc);
      acc = wmma_matvec(p.Whh_bf, p.h_bf, HID, r0, lane, acc);
      if ((lane & 15) == 0) {
        bf16* hnb = (t & 1) ? p.hnb1 : p.hnb0;
        const int dbase = d0 + ((lane >> 4) << 1);   // lane0: d0; lane16: d0+2
        #pragma unroll
        for (int q = 0; q < 2; ++q) {
          const int d = dbase + q;
          float ig = sigmoidf_(acc[4 * q + 0] + p.b_ih[d]           + p.b_hh[d]);
          float fg = sigmoidf_(acc[4 * q + 1] + p.b_ih[HID + d]     + p.b_hh[HID + d]);
          float gg = tanhf    (acc[4 * q + 2] + p.b_ih[2 * HID + d] + p.b_hh[2 * HID + d]);
          float og = sigmoidf_(acc[4 * q + 3] + p.b_ih[3 * HID + d] + p.b_hh[3 * HID + d]);
          float cn = fg * p.c[d] + ig * gg;
          float hn = og * tanhf(cn);
          p.c[d] = cn;
          p.h[d] = hn;
          hnb[d] = (bf16)hn;
        }
      }
    }
    grid_barrier(p.sync);

    // ---- P5: logits (WMMA, 2000 waves x 16 rows). No barrier: overlaps next P1. ----
    if (b < 250) {
      const bf16* hv = (t & 1) ? p.hnb1 : p.hnb0;   // double-buffered vs P3(t+1)
      int r0 = (b * 8 + wave) * 16;
      v8f acc = {};
      acc = wmma_matvec(p.Wout_bf, hv, HID, r0, lane, acc);
      if ((lane & 15) == 0) {
        const int rl = r0 + ((lane >> 4) << 3);
        float* outp = p.out_logits + (size_t)t * VOC + rl;
        const float4* bo = (const float4*)(p.b_out + rl);
        float4 t0 = bo[0], t1 = bo[1];
        float4 o0, o1;
        o0.x = acc[0] + t0.x; o0.y = acc[1] + t0.y;
        o0.z = acc[2] + t0.z; o0.w = acc[3] + t0.w;
        o1.x = acc[4] + t1.x; o1.y = acc[5] + t1.y;
        o1.z = acc[6] + t1.z; o1.w = acc[7] + t1.w;
        float4* oq = (float4*)outp;
        oq[0] = o0; oq[1] = o1;
      }
    }
  }
}

extern "C" void kernel_launch(void* const* d_in, const int* in_sizes, int n_in,
                              void* d_out, int out_size, void* d_ws, size_t ws_size,
                              hipStream_t stream) {
  Params p{};
  p.enc    = (const float*)d_in[0];
  p.h0     = (const float*)d_in[1];
  p.c0     = (const float*)d_in[2];
  p.sos    = (const int*)  d_in[3];
  p.tgt    = (const int*)  d_in[4];
  p.emb    = (const float*)d_in[5];
  p.Wa_enc = (const float*)d_in[6];
  p.Wa_h   = (const float*)d_in[7];
  p.v_a    = (const float*)d_in[8];
  p.W_ih   = (const float*)d_in[9];
  p.W_hh   = (const float*)d_in[10];
  p.b_ih   = (const float*)d_in[11];
  p.b_hh   = (const float*)d_in[12];
  p.W_out  = (const float*)d_in[13];
  p.b_out  = (const float*)d_in[14];
  p.out_logits = (float*)d_out;
  p.out_attn   = (float*)d_out + (size_t)T_LEN * VOC;

  char* ws = (char*)d_ws;
  size_t off = 0;
  auto take = [&](size_t bytes) -> char* {
    char* r = ws + off;
    off = (off + bytes + 255) & ~(size_t)255;
    return r;
  };
  p.sync     = (unsigned*)take(256);
  p.h        = (float*)take(HID * 4);
  p.c        = (float*)take(HID * 4);
  p.wexp     = (float*)take(S_LEN * 4);
  p.partialZ = (float*)take(32 * 4);
  p.enc_proj = (float*)take((size_t)S_LEN * ATT * 4);
  p.Wa_hT    = (float*)take((size_t)ATT * HID * 4);
  p.h_bf     = (bf16*)take(HID * 2);
  p.x_bf     = (bf16*)take(HID * 2);
  p.hnb0     = (bf16*)take(HID * 2);
  p.hnb1     = (bf16*)take(HID * 2);
  p.Wih_bf   = (bf16*)take((size_t)4 * HID * HID * 2);
  p.Whh_bf   = (bf16*)take((size_t)4 * HID * HID * 2);
  p.Wout_bf  = (bf16*)take((size_t)VOC * HID * 2);
  if (off > ws_size) return;  // workspace too small for bf16 weight cache

  hipMemsetAsync(p.sync, 0, 256, stream);  // graph-capture-safe barrier reset
  hipLaunchKernelGGL(decoder_persistent, dim3(NB), dim3(NTH), 0, stream, p);
}
